// Model_3315714753203
// MI455X (gfx1250) — matrix operand order repaired and checked by reading, not compiled, not run on hardware
//
#include <hip/hip_runtime.h>

// ---------------- problem constants ----------------
#define B_  4
#define S_  2048
#define I_  512
#define D_  512
#define H_  8
#define P_  720
#define HD_ 64

typedef _Float16 half_t;
typedef __attribute__((ext_vector_type(16))) _Float16 v16h;
typedef __attribute__((ext_vector_type(8)))  _Float16 v8h;
typedef __attribute__((ext_vector_type(8)))  float    v8f;

// D = A(16x32 f16) * B(32x16 f16) + C(16x16 f32), wave32 WMMA
__device__ __forceinline__ v8f wmma_f16(v16h a, v16h b, v8f c) {
  return __builtin_amdgcn_wmma_f32_16x16x32_f16(
      /*neg_a=*/false, a, /*neg_b=*/false, b,
      /*c_mod=*/(short)0, c, /*reuse_a=*/false, /*reuse_b=*/false);
}

// CDNA5 async copy: global -> LDS, 16B per lane, tracked by ASYNCcnt.
__device__ __forceinline__ void async_ld_b128(unsigned lds_byte_off, const void* gaddr) {
  asm volatile("global_load_async_to_lds_b128 %0, %1, off"
               :: "v"(lds_byte_off), "v"(gaddr) : "memory");
}
__device__ __forceinline__ void wait_async_le16() {
  asm volatile("s_wait_asynccnt 0x10" ::: "memory");
}
__device__ __forceinline__ void wait_async_0() {
  asm volatile("s_wait_asynccnt 0x0" ::: "memory");
}

// A-fragment loader: 16x32 tile, row-major source with leading dim `ld` (halves).
//   lane<16 : elems {0..15}->K0..15
//   lane>=16: elems {0..15}->K16..31
__device__ __forceinline__ v16h load_afrag(const half_t* row_base, int ld) {
  const int lane = threadIdx.x & 31;
  const int hh = lane >> 4;
  const int r  = lane & 15;
  const half_t* p = row_base + (size_t)r * ld;
  return *(const v16h*)(p + hh * 16);        // K = hh*16 .. hh*16+15
}

// B-fragment loader: 32x16 tile from "N-major" storage (row index = N, contiguous K),
// leading dim `ld` (halves). Lane n(=lane&15) holds column N=n;
// lanes 0..15 cover K=0..15, lanes 16..31 cover K=16..31 (contiguous per lane).
__device__ __forceinline__ v16h load_bfrag(const half_t* nrow_base, int ld) {
  const int lane = threadIdx.x & 31;
  const int hh = lane >> 4;
  const int n  = lane & 15;
  return *(const v16h*)(nrow_base + (size_t)n * ld + hh * 16);
}

// ---------------- prep kernels ----------------
__global__ void k_f32_to_f16(const float* __restrict__ src, half_t* __restrict__ dst, int n) {
  int i = blockIdx.x * blockDim.x + threadIdx.x;
  if (i < n) dst[i] = (half_t)src[i];
}

// W[H,I,HD] f32 -> WT[H,HD,I] f16 (so B-fragments read contiguous K=I)
__global__ void k_w_transpose(const float* __restrict__ src, half_t* __restrict__ dst) {
  int i = blockIdx.x * blockDim.x + threadIdx.x;   // over H_*I_*HD_
  int d  = i % HD_;
  int t  = i / HD_;
  int ii = t % I_;
  int h  = t / I_;
  dst[((size_t)h * HD_ + d) * I_ + ii] = (half_t)src[i];
}

// ---------------- QKV projection: q/k -> [B,H,S,HD] f16, v -> [B,H,HD,S] f16 ----------------
__global__ void k_qkv_proj(const half_t* __restrict__ Xh,
                           const half_t* __restrict__ WqT, const half_t* __restrict__ WkT,
                           const half_t* __restrict__ WvT,
                           const float* __restrict__ bq, const float* __restrict__ bk,
                           const float* __restrict__ bv,
                           half_t* __restrict__ qH, half_t* __restrict__ kH,
                           half_t* __restrict__ vTH) {
  const int s0 = blockIdx.x * 16;
  const int bh = blockIdx.y;             // b*H + h
  const int b  = bh / H_, h = bh % H_;
  const int which = blockIdx.z;          // 0=q 1=k 2=v
  const int wave  = threadIdx.x >> 5;    // 4 waves -> 4 hd col tiles
  const int n0    = wave * 16;
  const int lane  = threadIdx.x & 31;
  const int cl = lane & 15, ch = lane >> 4;

  const half_t* WT   = (which == 0) ? WqT : (which == 1) ? WkT : WvT;
  const float*  bias = (which == 0) ? bq  : (which == 1) ? bk  : bv;

  const half_t* Arow = Xh + ((size_t)(b * S_ + s0)) * I_;
  const half_t* Brow = WT + ((size_t)h * HD_ + n0) * I_;

  v8f acc = {};
#pragma unroll 4
  for (int k0 = 0; k0 < I_; k0 += 32)
    acc = wmma_f16(load_afrag(Arow + k0, I_), load_bfrag(Brow + k0, I_), acc);

  const float bi = bias[h * HD_ + n0 + cl];
  if (which == 2) {  // v, transposed [HD,S]
    half_t* outp = vTH + ((size_t)bh * HD_ + n0 + cl) * S_ + s0;
#pragma unroll
    for (int v = 0; v < 8; ++v) outp[v + 8 * ch] = (half_t)(acc[v] + bi);
  } else {           // q/k, [S,HD]
    half_t* outp = ((which == 0) ? qH : kH) + ((size_t)bh * S_ + s0) * HD_ + n0 + cl;
#pragma unroll
    for (int v = 0; v < 8; ++v) outp[(size_t)(v + 8 * ch) * HD_] = (half_t)(acc[v] + bi);
  }
}

// ---------------- flash attention: one wave per (b,h, 16-row q tile) ----------------
// K/V tiles are double-buffered in LDS via CDNA5 async global->LDS copies (ASYNCcnt),
// so the next tile streams in while the current tile's WMMAs + softmax execute.
__global__ void k_attention(const half_t* __restrict__ qH, const half_t* __restrict__ kH,
                            const half_t* __restrict__ vTH, half_t* __restrict__ ctxT) {
  __shared__ __align__(32) half_t Kt[2][32 * HD_];   // 2 x 4KB : rows=t(32), ld=HD
  __shared__ __align__(32) half_t Vt[2][HD_ * 32];   // 2 x 4KB : rows=hd(64), ld=32
  __shared__ __align__(32) half_t Plds[16 * 32];     // 1KB probability tile
  const int s0 = blockIdx.x * 16;
  const int bh = blockIdx.y;
  const int b = bh / H_, h = bh % H_;
  const int lane = threadIdx.x & 31;
  const int cl = lane & 15, ch = lane >> 4;

  const half_t* qrow = qH + ((size_t)bh * S_ + s0) * HD_;
  const v16h qa0 = load_afrag(qrow, HD_);        // K(hd)=0..31
  const v16h qa1 = load_afrag(qrow + 32, HD_);   // K(hd)=32..63

  float m_i[8], l_i[8];
  v8f acc[4] = {};                               // 16 x 64 ctx accumulator
#pragma unroll
  for (int v = 0; v < 8; ++v) { m_i[v] = -1e30f; l_i[v] = 0.0f; }

  const half_t* kbase = kH + (size_t)bh * S_ * HD_;
  const half_t* vbase = vTH + (size_t)bh * HD_ * S_;
  const float scale = 1.0f / 64.0f;              // reference divides by hd, not sqrt(hd)

  const unsigned kbuf0 = (unsigned)(size_t)(void*)&Kt[0][0];
  const unsigned vbuf0 = (unsigned)(size_t)(void*)&Vt[0][0];

  // issue one 32-column K/V tile (8 + 8 lane-wide b128 async transfers)
  auto issue_tile = [&](int t0, int db) {
    const char* kg = (const char*)(kbase + (size_t)t0 * HD_);   // contiguous 4KB
    const unsigned klds = kbuf0 + (unsigned)db * (32 * HD_ * 2);
#pragma unroll
    for (int j = 0; j < 8; ++j) {
      const int idx = j * 32 + lane;                            // 0..255 16B chunks
      async_ld_b128(klds + idx * 16, kg + idx * 16);
    }
    const half_t* vg = vbase + t0;                              // 64 rows x 64B, stride S
    const unsigned vlds = vbuf0 + (unsigned)db * (HD_ * 32 * 2);
#pragma unroll
    for (int j = 0; j < 8; ++j) {
      const int idx = j * 32 + lane;
      const int row = idx >> 2, c4 = idx & 3;                   // row 0..63, 16B chunk 0..3
      async_ld_b128(vlds + row * 64 + c4 * 16,
                    (const char*)(vg + (size_t)row * S_) + c4 * 16);
    }
  };

  issue_tile(0, 0);
  for (int t0 = 0; t0 < S_; t0 += 32) {
    const int db = (t0 >> 5) & 1;
    if (t0 + 32 < S_) {
      issue_tile(t0 + 32, db ^ 1);   // prefetch next tile into other buffer
      wait_async_le16();             // current tile's 16 transfers retired (in-order)
    } else {
      wait_async_0();
    }
    const half_t* ktile = &Kt[db][0];
    const half_t* vtile = &Vt[db][0];

    v8f sc[2];
#pragma unroll
    for (int tt = 0; tt < 2; ++tt) {
      const half_t* kb = ktile + (size_t)(tt * 16) * HD_;       // rows = N(t), contiguous K(hd)
      v8f s = {};
      s = wmma_f16(qa0, load_bfrag(kb, HD_), s);
      s = wmma_f16(qa1, load_bfrag(kb + 32, HD_), s);
      sc[tt] = s;
    }
    // online softmax per row (row = v + 8*ch; columns live across the 16-lane half)
#pragma unroll
    for (int v = 0; v < 8; ++v) {
      float s0v = sc[0][v] * scale;
      float s1v = sc[1][v] * scale;
      float tm = fmaxf(s0v, s1v);
#pragma unroll
      for (int x = 1; x <= 8; x <<= 1) tm = fmaxf(tm, __shfl_xor(tm, x, 32));
      float nm    = fmaxf(m_i[v], tm);
      float alpha = __expf(m_i[v] - nm);
      float p0 = __expf(s0v - nm);
      float p1 = __expf(s1v - nm);
      float rs = p0 + p1;
#pragma unroll
      for (int x = 1; x <= 8; x <<= 1) rs += __shfl_xor(rs, x, 32);
      l_i[v] = l_i[v] * alpha + rs;
      m_i[v] = nm;
#pragma unroll
      for (int c = 0; c < 4; ++c) acc[c][v] *= alpha;
      Plds[(v + 8 * ch) * 32 + cl]      = (half_t)p0;   // cols t0..t0+15
      Plds[(v + 8 * ch) * 32 + 16 + cl] = (half_t)p1;   // cols t0+16..t0+31
    }
    __syncthreads();
    const v16h pa = load_afrag(Plds, 32);               // P as A-fragment (K = t local)
#pragma unroll
    for (int c = 0; c < 4; ++c) {
      const half_t* vb = vtile + (size_t)(c * 16) * 32; // rows = N(hd), contiguous K(t), ld=32
      acc[c] = wmma_f16(pa, load_bfrag(vb, 32), acc[c]);
    }
    __syncthreads();
  }

  // normalize and write ctx transposed: ctxT[b][d = h*64+hd][s]  (B operand of temporal GEMM)
#pragma unroll
  for (int v = 0; v < 8; ++v) {
    float inv = 1.0f / l_i[v];
#pragma unroll
    for (int c = 0; c < 4; ++c) {
      ctxT[((size_t)b * D_ + h * HD_ + c * 16 + cl) * S_ + s0 + v + 8 * ch] =
          (half_t)(acc[c][v] * inv);
    }
  }
}

// ---------------- temporal projection: up[b,p,d] = sum_s Wt[p,s]*ctxT[b,d,s] + bt[p] ----------------
__global__ void k_temporal(const half_t* __restrict__ WtH, const half_t* __restrict__ ctxT,
                           const float* __restrict__ bt, half_t* __restrict__ upH) {
  const int p0 = blockIdx.x * 16;
  const int b  = blockIdx.z;
  const int wave = threadIdx.x >> 5;
  const int d0 = blockIdx.y * 64 + wave * 16;
  const int lane = threadIdx.x & 31;
  const int cl = lane & 15, ch = lane >> 4;

  const half_t* Arow = WtH + (size_t)p0 * S_;                  // rows = p, contiguous K=s
  const half_t* Brow = ctxT + ((size_t)b * D_ + d0) * S_;      // rows = N(d), contiguous K=s
  v8f acc = {};
  for (int k0 = 0; k0 < S_; k0 += 32)
    acc = wmma_f16(load_afrag(Arow + k0, S_), load_bfrag(Brow + k0, S_), acc);

  half_t* outp = upH + ((size_t)b * P_ + p0) * D_ + d0 + cl;
#pragma unroll
  for (int v = 0; v < 8; ++v)
    outp[(size_t)(v + 8 * ch) * D_] = (half_t)(acc[v] + bt[p0 + v + 8 * ch]);
}

// ---------------- output projection: out[b,p,o] = sum_d up[b,p,d]*Wo[o,d] + bo[o] (f32 out) ----------------
__global__ void k_output(const half_t* __restrict__ upH, const half_t* __restrict__ WoH,
                         const float* __restrict__ bo, float* __restrict__ out) {
  const int p0 = blockIdx.x * 16;
  const int b  = blockIdx.z;
  const int wave = threadIdx.x >> 5;
  const int o0 = blockIdx.y * 64 + wave * 16;
  const int lane = threadIdx.x & 31;
  const int cl = lane & 15, ch = lane >> 4;

  const half_t* Arow = upH + ((size_t)b * P_ + p0) * D_;   // rows = p, contiguous K=d
  const half_t* Brow = WoH + (size_t)o0 * D_;              // rows = N(o), contiguous K=d
  v8f acc = {};
#pragma unroll 4
  for (int k0 = 0; k0 < D_; k0 += 32)
    acc = wmma_f16(load_afrag(Arow + k0, D_), load_bfrag(Brow + k0, D_), acc);

  float* op = out + ((size_t)b * P_ + p0) * I_ + o0 + cl;
  const float bi = bo[o0 + cl];
#pragma unroll
  for (int v = 0; v < 8; ++v)
    op[(size_t)(v + 8 * ch) * I_] = acc[v] + bi;
}

// ---------------- host launch ----------------
extern "C" void kernel_launch(void* const* d_in, const int* in_sizes, int n_in,
                              void* d_out, int out_size, void* d_ws, size_t ws_size,
                              hipStream_t stream) {
  (void)in_sizes; (void)n_in; (void)out_size; (void)ws_size;
  const float* inp = (const float*)d_in[0];
  const float* Wq  = (const float*)d_in[1];
  const float* bq  = (const float*)d_in[2];
  const float* Wk  = (const float*)d_in[3];
  const float* bk  = (const float*)d_in[4];
  const float* Wv  = (const float*)d_in[5];
  const float* bv  = (const float*)d_in[6];
  const float* Wt  = (const float*)d_in[7];
  const float* bt  = (const float*)d_in[8];
  const float* Wo  = (const float*)d_in[9];
  const float* bo  = (const float*)d_in[10];
  float* out = (float*)d_out;

  char* ws = (char*)d_ws;
  size_t off = 0;
  auto take = [&](size_t nelem) -> half_t* {
    half_t* p = (half_t*)(ws + off);
    off += (nelem * sizeof(half_t) + 255) & ~(size_t)255;
    return p;
  };
  half_t* Xh   = take((size_t)B_ * S_ * I_);
  half_t* WqT  = take((size_t)H_ * HD_ * I_);
  half_t* WkT  = take((size_t)H_ * HD_ * I_);
  half_t* WvT  = take((size_t)H_ * HD_ * I_);
  half_t* WtH  = take((size_t)P_ * S_);
  half_t* WoH  = take((size_t)I_ * D_);
  half_t* qHp  = take((size_t)B_ * H_ * S_ * HD_);
  half_t* kHp  = take((size_t)B_ * H_ * S_ * HD_);
  half_t* vTH  = take((size_t)B_ * H_ * HD_ * S_);
  half_t* ctxT = take((size_t)B_ * D_ * S_);
  half_t* upH  = take((size_t)B_ * P_ * D_);

  const int nX = B_ * S_ * I_;
  k_f32_to_f16<<<(nX + 255) / 256, 256, 0, stream>>>(inp, Xh, nX);
  const int nW = H_ * I_ * HD_;
  k_w_transpose<<<(nW + 255) / 256, 256, 0, stream>>>(Wq, WqT);
  k_w_transpose<<<(nW + 255) / 256, 256, 0, stream>>>(Wk, WkT);
  k_w_transpose<<<(nW + 255) / 256, 256, 0, stream>>>(Wv, WvT);
  const int nWt = P_ * S_;
  k_f32_to_f16<<<(nWt + 255) / 256, 256, 0, stream>>>(Wt, WtH, nWt);
  const int nWo = I_ * D_;
  k_f32_to_f16<<<(nWo + 255) / 256, 256, 0, stream>>>(Wo, WoH, nWo);

  k_qkv_proj<<<dim3(S_ / 16, B_ * H_, 3), 128, 0, stream>>>(
      Xh, WqT, WkT, WvT, bq, bk, bv, qHp, kHp, vTH);
  k_attention<<<dim3(S_ / 16, B_ * H_), 32, 0, stream>>>(qHp, kHp, vTH, ctxT);
  k_temporal<<<dim3(P_ / 16, D_ / 64, B_), 128, 0, stream>>>(WtH, ctxT, bt, upH);
  k_output<<<dim3(P_ / 16, I_ / 64, B_), 128, 0, stream>>>(upH, WoH, bo, out);
}
